// RotaryMultiHeadAttention_11768210391692
// MI455X (gfx1250) — matrix operand order repaired and checked by reading, not compiled, not run on hardware
//
#include <hip/hip_runtime.h>
#include <hip/hip_bf16.h>
#include <math.h>

namespace {

constexpr int kB  = 2;
constexpr int kS  = 2048;
constexpr int kD  = 2048;
constexpr int kH  = 16;
constexpr int kHD = 128;
constexpr int kM  = kB * kS;                       // 4096 rows of X
constexpr float kScale      = 0.08838834764831845f; // 1/sqrt(128)
constexpr float kLog2e      = 1.4426950408889634f;
constexpr float kLog2_10000 = 13.287712379549449f;
constexpr float kNegInf     = -3.0e38f;

typedef __attribute__((ext_vector_type(16))) __bf16 v16bf;
typedef __attribute__((ext_vector_type(8)))  __bf16 v8bf;
typedef __attribute__((ext_vector_type(4)))  __bf16 v4bf;
typedef __attribute__((ext_vector_type(8)))  float  v8f;
typedef __attribute__((ext_vector_type(4)))  int    v4i;

// ---- gfx1250 async global->LDS copy (guarded; falls back to ld+ds_store) ----
#if __has_builtin(__builtin_amdgcn_global_load_async_to_lds_b128) && \
    __has_builtin(__builtin_amdgcn_s_wait_asynccnt)
#define USE_ASYNC_COPY 1
typedef __attribute__((address_space(1))) v4i as1_v4i;  // global int4
typedef __attribute__((address_space(3))) v4i as3_v4i;  // LDS int4
#else
#define USE_ASYNC_COPY 0
#endif

__device__ __forceinline__ void copy16_g2lds(const __bf16* g, __bf16* l) {
#if USE_ASYNC_COPY
  __builtin_amdgcn_global_load_async_to_lds_b128(
      (as1_v4i*)(uintptr_t)g, (as3_v4i*)(uintptr_t)l, 0, 0);
#else
  *(v8bf*)l = *(const v8bf*)g;
#endif
}

__device__ __forceinline__ void wait_async_copies() {
#if USE_ASYNC_COPY
  __builtin_amdgcn_s_wait_asynccnt(0);
#endif
}

__device__ __forceinline__ v8f zero8() {
  v8f z = {0.f, 0.f, 0.f, 0.f, 0.f, 0.f, 0.f, 0.f};
  return z;
}

__device__ __forceinline__ v8f wmma_bf16(v16bf a, v16bf b, v8f c) {
  return __builtin_amdgcn_wmma_f32_16x16x32_bf16(
      /*neg_a=*/false, a, /*neg_b=*/false, b,
      /*c_mod=*/(short)0, c, /*reuse_a=*/false, /*reuse_b=*/false);
}

// A fragment: 16x32 bf16 (row-major, leading dim ld). Layout:
// element e -> K = e + (lane<16?0:16).
__device__ __forceinline__ v16bf load_a16x32(const __bf16* tile, int ld) {
  const int lane = threadIdx.x & 31;
  const int row  = lane & 15;
  const int koff = (lane < 16) ? 0 : 16;
  const __bf16* p = tile + (size_t)row * ld + koff;
  v8bf lo = *(const v8bf*)(p);
  v8bf hi = *(const v8bf*)(p + 8);
  v16bf a;
#pragma unroll
  for (int i = 0; i < 8; ++i) { a[i] = lo[i]; a[i + 8] = hi[i]; }
  return a;
}

// B fragment: lane holds column n=lane&15; element e -> K = e + (lane<16?0:16).
// p must point at the 16 contiguous K-values for this lane.
__device__ __forceinline__ v16bf load_b_rows(const __bf16* p) {
  v8bf lo = *(const v8bf*)(p);
  v8bf hi = *(const v8bf*)(p + 8);
  v16bf b;
#pragma unroll
  for (int i = 0; i < 8; ++i) { b[i] = lo[i]; b[i + 8] = hi[i]; }
  return b;
}

// ---------------------------------------------------------------- cast kernel
__global__ void cvt_f32_to_bf16(const float* __restrict__ src,
                                __bf16* __restrict__ dst, long n4) {
  long i = (long)blockIdx.x * blockDim.x + threadIdx.x;
  const long stride = (long)gridDim.x * blockDim.x;
  for (; i < n4; i += stride) {
    float4 f = ((const float4*)src)[i];
    v4bf o;
    o[0] = (__bf16)f.x; o[1] = (__bf16)f.y;
    o[2] = (__bf16)f.z; o[3] = (__bf16)f.w;
    ((v4bf*)dst)[i] = o;
  }
}

// -------------------------------------------------------------- WMMA GEMM
// Y[m,n] = sum_k X[m,k] * W[n,k]   (X: kM x kD bf16, W: kD x kD bf16 row-major)
// X 64x32 slab is staged through LDS (async, double-buffered) and shared by
// all 8 waves of the block.
// MODE 0: bf16 out, (B,H,S,HD) scatter (V projection)
// MODE 1: MODE 0 + fused RoPE (Q/K projections)
// MODE 2: fp32 out, flat (B,S,D) (output projection)
template <int MODE>
__global__ __launch_bounds__(256) void gemm_wmma_kernel(
    const __bf16* __restrict__ X, const __bf16* __restrict__ W,
    __bf16* __restrict__ Yh, float* __restrict__ Yf) {
  __shared__ __align__(16) __bf16 xtile[2][64 * 40];  // 64 rows x 32 k (+pad)

  const int lane = threadIdx.x & 31;
  const int wave = threadIdx.x >> 5;
  const int mblk = blockIdx.x * 64;
  const int m0 = mblk + (wave >> 2) * 32;
  const int n0 = blockIdx.y * 128 + (wave & 3) * 32;

  const int bn  = lane & 15;
  const int bkb = (lane < 16) ? 0 : 16;

  // per-thread staging chunk: 256 threads x 16B cover the 64x32 slab
  const int crow = threadIdx.x >> 2;        // 0..63
  const int ccol = (threadIdx.x & 3) * 8;   // 0,8,16,24
  const __bf16* xsrc = X + (size_t)(mblk + crow) * kD + ccol;
  __bf16* xdst0 = &xtile[0][crow * 40 + ccol];
  __bf16* xdst1 = &xtile[1][crow * 40 + ccol];

  v8f acc00 = zero8(), acc01 = zero8(), acc10 = zero8(), acc11 = zero8();

  copy16_g2lds(xsrc, xdst0);                // prime the pipeline (k0 = 0)

  const int awr = (wave >> 2) * 32;         // wave's row offset in the slab
  for (int k0 = 0; k0 < kD; k0 += 32) {
    const int cur = (k0 >> 5) & 1;
    wait_async_copies();
    __syncthreads();                        // slab(cur) ready for everyone
    if (k0 + 32 < kD)                       // overlap next copy with compute
      copy16_g2lds(xsrc + k0 + 32, cur ? xdst0 : xdst1);

    v16bf a0 = load_a16x32(&xtile[cur][awr * 40], 40);
    v16bf a1 = load_a16x32(&xtile[cur][(awr + 16) * 40], 40);
    v16bf b0 = load_b_rows(W + (size_t)(n0 + bn) * kD + k0 + bkb);
    v16bf b1 = load_b_rows(W + (size_t)(n0 + 16 + bn) * kD + k0 + bkb);
    acc00 = wmma_bf16(a0, b0, acc00);
    acc01 = wmma_bf16(a0, b1, acc01);
    acc10 = wmma_bf16(a1, b0, acc10);
    acc11 = wmma_bf16(a1, b1, acc11);
    __builtin_prefetch(W + (size_t)(n0 + bn) * kD + k0 + 32 + bkb, 0, 3);
  }

  const int rsel = (lane < 16) ? 0 : 8;
  v8f accs[2][2] = {{acc00, acc01}, {acc10, acc11}};
#pragma unroll
  for (int mi = 0; mi < 2; ++mi) {
#pragma unroll
    for (int ni = 0; ni < 2; ++ni) {
      const int colg = n0 + ni * 16 + (lane & 15);
#pragma unroll
      for (int v = 0; v < 8; ++v) {
        const int rowg = m0 + mi * 16 + rsel + v;
        float val = accs[mi][ni][v];
        if (MODE == 2) {
          Yf[(size_t)rowg * kD + colg] = val;
        } else {
          const int bidx = rowg >> 11;        // / S
          const int srow = rowg & (kS - 1);
          const int h = colg >> 7;            // / HD
          const int j = colg & (kHD - 1);
          if (MODE == 1) {
            // adjacent head-dim columns live in lane ^ 1
            float part = __shfl_xor(val, 1, 32);
            float rot  = (j & 1) ? part : -part;
            float invf = exp2f(-(float)(j & ~1) * (kLog2_10000 / 128.0f));
            float ang  = (float)srow * invf;
            float sn, cs;
            sincosf(ang, &sn, &cs);
            val = val * cs + rot * sn;
          }
          Yh[(((size_t)bidx * kH + h) * kS + srow) * kHD + j] = (__bf16)val;
        }
      }
    }
  }
}

// --------------------------------------------------------- flash attention
// Transposed formulation: S^T = K * Q^T, O^T = V^T * P^T.
// Query index == WMMA column == lane -> softmax stats are per-lane scalars.
// Block: 128 threads (4 waves), 64 queries per block (16 per wave).
__global__ __launch_bounds__(128) void flash_attn_kernel(
    const __bf16* __restrict__ Q, const __bf16* __restrict__ K,
    const __bf16* __restrict__ V, __bf16* __restrict__ O) {
  __shared__ __align__(16) __bf16 ktile[32 * 136];   // 32 keys x 128 hd (+pad)
  __shared__ __align__(16) __bf16 vtile[128 * 40];   // 128 hd x 32 keys (+pad)

  const int tid  = threadIdx.x;
  const int lane = tid & 31;
  const int wave = tid >> 5;
  const int bh   = blockIdx.y;          // b*16 + h
  const int b    = bh >> 4;
  const int h    = bh & 15;
  const int qt0  = blockIdx.x * 64;
  const int q0w  = qt0 + wave * 16;

  const size_t bhbase = (size_t)bh * kS * kHD;

  // Q^T B-fragments (n = query, K-dim = head dim, 4 slices of 32)
  const int qn = lane & 15;
  const int kb = (lane < 16) ? 0 : 16;
  v16bf qb[4];
  {
    const __bf16* qrow = Q + bhbase + (size_t)(q0w + qn) * kHD;
#pragma unroll
    for (int f = 0; f < 4; ++f) {
      const __bf16* p = qrow + f * 32 + kb;
      v8bf lo = *(const v8bf*)p;
      v8bf hi = *(const v8bf*)(p + 8);
#pragma unroll
      for (int i = 0; i < 8; ++i) { qb[f][i] = lo[i]; qb[f][i + 8] = hi[i]; }
    }
  }

  v8f oacc[8];
#pragma unroll
  for (int t = 0; t < 8; ++t) oacc[t] = zero8();
  float mrow = kNegInf;
  float lrow = 0.0f;

  const int nkt = (qt0 >> 5) + 2;       // key tiles of 32, causal coverage
  for (int kt = 0; kt < nkt; ++kt) {
    const int kt0 = kt * 32;
    // cooperative staging: K via async copy, V transposed through VGPRs
    {
      const __bf16* kbase = K + bhbase + (size_t)kt0 * kHD;
      const __bf16* vbase = V + bhbase + (size_t)kt0 * kHD;
#pragma unroll
      for (int it = 0; it < 4; ++it) {
        int u   = tid + it * 128;       // 0..511 chunks of 8 bf16
        int key = u >> 4;
        int hdc = (u & 15) * 8;
        copy16_g2lds(kbase + key * kHD + hdc, &ktile[key * 136 + hdc]);
        v8bf vv = *(const v8bf*)(vbase + key * kHD + hdc);
#pragma unroll
        for (int i = 0; i < 8; ++i) vtile[(hdc + i) * 40 + key] = vv[i];
      }
    }
    wait_async_copies();
    __syncthreads();

    // S^T tiles: keys 0-15 (st0) and 16-31 (st1) x 16 queries
    v8f st0 = zero8(), st1 = zero8();
    {
      const int krow = lane & 15;
      const int koff = (lane < 16) ? 0 : 16;
#pragma unroll
      for (int f = 0; f < 4; ++f) {
        const __bf16* p0 = &ktile[krow * 136 + f * 32 + koff];
        const __bf16* p1 = &ktile[(krow + 16) * 136 + f * 32 + koff];
        v8bf l0 = *(const v8bf*)p0, h0 = *(const v8bf*)(p0 + 8);
        v8bf l1 = *(const v8bf*)p1, h1 = *(const v8bf*)(p1 + 8);
        v16bf a0, a1;
#pragma unroll
        for (int i = 0; i < 8; ++i) {
          a0[i] = l0[i]; a0[i + 8] = h0[i];
          a1[i] = l1[i]; a1[i + 8] = h1[i];
        }
        st0 = wmma_bf16(a0, qb[f], st0);
        st1 = wmma_bf16(a1, qb[f], st1);
      }
    }

    // causal mask + scale + online softmax (per-lane query stats)
    const int q    = q0w + (lane & 15);
    const int ksel = (lane < 16) ? 0 : 8;
    float s0[8], s1[8];
    float mt = kNegInf;
#pragma unroll
    for (int v = 0; v < 8; ++v) {
      const int key0 = kt0 + ksel + v;
      const int key1 = key0 + 16;
      s0[v] = (key0 <= q) ? st0[v] * kScale : kNegInf;
      s1[v] = (key1 <= q) ? st1[v] * kScale : kNegInf;
      mt = fmaxf(mt, fmaxf(s0[v], s1[v]));
    }
    mt = fmaxf(mt, __shfl_xor(mt, 16, 32));
    const float mnew  = fmaxf(mrow, mt);
    const float alpha = exp2f((mrow - mnew) * kLog2e);
    mrow = mnew;

    float p0[8], p1[8];
    float rs = 0.0f;
#pragma unroll
    for (int v = 0; v < 8; ++v) {
      p0[v] = exp2f((s0[v] - mnew) * kLog2e);
      p1[v] = exp2f((s1[v] - mnew) * kLog2e);
      rs += p0[v] + p1[v];
    }
    rs += __shfl_xor(rs, 16, 32);
    lrow = lrow * alpha + rs;

    // P^T B-fragment straight from S^T accumulators (half-wave swap only)
    v16bf pb;
#pragma unroll
    for (int v = 0; v < 8; ++v) {
      const float x0 = __shfl_xor(p0[v], 16, 32);
      const float x1 = __shfl_xor(p1[v], 16, 32);
      pb[v]     = (__bf16)((lane < 16) ? p0[v] : x1);
      pb[v + 8] = (__bf16)((lane < 16) ? x0 : p1[v]);
    }

    // rescale running O^T and accumulate V^T * P^T
#pragma unroll
    for (int t = 0; t < 8; ++t) {
#pragma unroll
      for (int v = 0; v < 8; ++v) oacc[t][v] *= alpha;
    }
    {
      const int hdl  = lane & 15;
      const int koff = (lane < 16) ? 0 : 16;
#pragma unroll
      for (int t = 0; t < 8; ++t) {
        const __bf16* p = &vtile[(t * 16 + hdl) * 40 + koff];
        v8bf lo = *(const v8bf*)p;
        v8bf hi = *(const v8bf*)(p + 8);
        v16bf va;
#pragma unroll
        for (int i = 0; i < 8; ++i) { va[i] = lo[i]; va[i + 8] = hi[i]; }
        oacc[t] = wmma_bf16(va, pb, oacc[t]);
      }
    }
    __syncthreads();
  }

  // epilogue: O^T -> (B,S,H*HD) bf16, normalized
  const float linv = 1.0f / lrow;
  const int   hsel = (lane < 16) ? 0 : 8;
  const int   srow = q0w + (lane & 15);
  __bf16* orow = O + ((size_t)b * kS + srow) * kD + h * kHD;
#pragma unroll
  for (int t = 0; t < 8; ++t) {
#pragma unroll
    for (int v = 0; v < 8; ++v)
      orow[t * 16 + hsel + v] = (__bf16)(oacc[t][v] * linv);
  }
}

}  // namespace

extern "C" void kernel_launch(void* const* d_in, const int* in_sizes, int n_in,
                              void* d_out, int out_size, void* d_ws, size_t ws_size,
                              hipStream_t stream) {
  (void)in_sizes; (void)n_in; (void)out_size; (void)ws_size;
  const float* x  = (const float*)d_in[0];
  const float* Wq = (const float*)d_in[1];
  const float* Wk = (const float*)d_in[2];
  const float* Wv = (const float*)d_in[3];
  const float* Wo = (const float*)d_in[4];

  const size_t NX = (size_t)kM * kD;            // 8388608
  const size_t NW = (size_t)kD * kD;            // 4194304
  const size_t NH = (size_t)kB * kH * kS * kHD; // 8388608

  __bf16* xb  = (__bf16*)d_ws;
  __bf16* wqb = xb  + NX;
  __bf16* wkb = wqb + NW;
  __bf16* wvb = wkb + NW;
  __bf16* wob = wvb + NW;
  __bf16* Qb  = wob + NW;
  __bf16* Kb  = Qb  + NH;
  __bf16* Vb  = Kb  + NH;
  __bf16* Ob  = Vb  + NH;

  // 1) fp32 -> bf16 casts
  cvt_f32_to_bf16<<<2048, 256, 0, stream>>>(x,  xb,  (long)(NX / 4));
  cvt_f32_to_bf16<<<1024, 256, 0, stream>>>(Wq, wqb, (long)(NW / 4));
  cvt_f32_to_bf16<<<1024, 256, 0, stream>>>(Wk, wkb, (long)(NW / 4));
  cvt_f32_to_bf16<<<1024, 256, 0, stream>>>(Wv, wvb, (long)(NW / 4));
  cvt_f32_to_bf16<<<1024, 256, 0, stream>>>(Wo, wob, (long)(NW / 4));

  // 2) QKV projections (RoPE fused into Q/K epilogues)
  dim3 ggrid(kM / 64, kD / 128);
  gemm_wmma_kernel<1><<<ggrid, 256, 0, stream>>>(xb, wqb, Qb, nullptr);
  gemm_wmma_kernel<1><<<ggrid, 256, 0, stream>>>(xb, wkb, Kb, nullptr);
  gemm_wmma_kernel<0><<<ggrid, 256, 0, stream>>>(xb, wvb, Vb, nullptr);

  // 3) causal flash attention
  flash_attn_kernel<<<dim3(kS / 64, kB * kH), 128, 0, stream>>>(Qb, Kb, Vb, Ob);

  // 4) output projection -> fp32 d_out
  gemm_wmma_kernel<2><<<ggrid, 256, 0, stream>>>(Ob, wob, nullptr, (float*)d_out);
}
